// WeightedPlackettLuceLoss_36189394437088
// MI455X (gfx1250) — compile-verified, matrix operand order checked by reading, not run on hardware
//
#include <hip/hip_runtime.h>
#include <hip/hip_bf16.h>

// Weighted Plackett-Luce loss, MI455X (gfx1250, wave32).
// B races x H=18 horses x P=3 score columns. Bandwidth-bound streaming
// reduction; WMMA used as the in-wave reduction engine.

#define H_CONST 18
#define P_CONST 3

typedef float    v2f  __attribute__((ext_vector_type(2)));
typedef float    v8f  __attribute__((ext_vector_type(8)));
typedef _Float16 v16h __attribute__((ext_vector_type(16)));

// Exact f32 wave32 sum via V_WMMA_F32_16X16X4_F32 (A = per-lane value in K=0/K=2
// slot, B = all ones -> D[m,n] = v_m + v_{m+16}; lane0 rows 0..7, lane16 rows 8..15).
__device__ __forceinline__ float wave_sum_f32(float v) {
#if __has_builtin(__builtin_amdgcn_wmma_f32_16x16x4_f32)
  v2f a; a.x = v;     a.y = 0.0f;
  v2f o; o.x = 1.0f;  o.y = 1.0f;
  v8f c = {0.f, 0.f, 0.f, 0.f, 0.f, 0.f, 0.f, 0.f};
  c = __builtin_amdgcn_wmma_f32_16x16x4_f32(false, a, false, o, (short)0, c,
                                            false, false);
  float s = c[0] + c[1] + c[2] + c[3] + c[4] + c[5] + c[6] + c[7];
  s += __shfl_xor(s, 16, 32);
  return s;
#else
  #pragma unroll
  for (int off = 16; off > 0; off >>= 1) v += __shfl_xor(v, off, 32);
  return v;
#endif
}

// Exact wave32 count (values are 0.0/1.0, representable in f16; accum in f32)
// via codegen-confirmed V_WMMA_F32_16X16X32_F16.
__device__ __forceinline__ float wave_count_wmma(float v01) {
  v16h a = {};
  a[0] = (_Float16)v01;
  v16h o;
  #pragma unroll
  for (int i = 0; i < 16; ++i) o[i] = (_Float16)1.0f;
  v8f c = {0.f, 0.f, 0.f, 0.f, 0.f, 0.f, 0.f, 0.f};
  c = __builtin_amdgcn_wmma_f32_16x16x32_f16(false, a, false, o, (short)0, c,
                                             false, false);
  float s = c[0] + c[1] + c[2] + c[3] + c[4] + c[5] + c[6] + c[7];
  s += __shfl_xor(s, 16, 32);
  return s;
}

__global__ void ws_zero_kernel(float* ws) {
  ws[0] = 0.0f;
  ws[1] = 0.0f;
}

__global__ void __launch_bounds__(256)
pl_loss_kernel(const float* __restrict__ scores,
               const int* __restrict__ rankings,
               const unsigned char* __restrict__ mask,
               float* __restrict__ ws, int B) {
  const int b   = blockIdx.x * 256 + threadIdx.x;
  const bool inr = (b < B);
  const long bc   = inr ? (long)b : 0;       // clamp: keep all lanes converged
  const long base = bc * H_CONST;
  const float* sp = scores + base * P_CONST;

  // Pull this race's 216B score block toward the caches before the gather.
  __builtin_prefetch(sp, 0, 0);
  __builtin_prefetch(sp + 32, 0, 0);

  int r[H_CONST];
  unsigned char mk[H_CONST];
  #pragma unroll
  for (int h = 0; h < H_CONST; ++h) r[h] = rankings[base + h];
  #pragma unroll
  for (int h = 0; h < H_CONST; ++h) mk[h] = mask[base + h];

  // First (stable) horse for each rank value 1..3.
  int h1 = -1, h2 = -1, h3 = -1;
  #pragma unroll
  for (int h = 0; h < H_CONST; ++h) {
    const bool ok = (mk[h] != 0) && (r[h] >= 1) && (r[h] <= P_CONST);
    if (ok && r[h] == 1 && h1 < 0) h1 = h;
    if (ok && r[h] == 2 && h2 < 0) h2 = h;
    if (ok && r[h] == 3 && h3 < 0) h3 = h;
  }
  const bool p1 = h1 >= 0, p2 = h2 >= 0, p3 = h3 >= 0;
  const int counts = (int)p1 + (int)p2 + (int)p3;
  // Horses at stable sorted positions 0 and 1 (position 2 == h3 when present).
  const int g0 = p1 ? h1 : (p2 ? h2 : h3);
  const int g1 = p1 ? (p2 ? h2 : h3) : h3;

  float loss = 0.0f;
  if (counts >= 2) {                 // counts<=1 -> all terms are exactly 0
    const float a0 = sp[g0 * P_CONST + 0];
    const float b0 = sp[g1 * P_CONST + 0];
    const float b1 = sp[g1 * P_CONST + 1];
    if (counts >= 3) {
      const float c0 = sp[h3 * P_CONST + 0];
      const float c1 = sp[h3 * P_CONST + 1];
      const float m0 = fmaxf(a0, fmaxf(b0, c0));
      const float l0 = m0 + logf(expf(a0 - m0) + expf(b0 - m0) + expf(c0 - m0));
      const float m1 = fmaxf(b1, c1);
      const float l1 = m1 + logf(expf(b1 - m1) + expf(c1 - m1));
      loss = (l0 - a0) + 0.8f * (l1 - b1);   // p=2 term is exactly 0
    } else {
      const float m0 = fmaxf(a0, b0);
      const float l0 = m0 + logf(expf(a0 - m0) + expf(b0 - m0));
      loss = l0 - a0;                        // p=1 single-member term is 0
    }
  }
  const float vld = (inr && counts > 0) ? 1.0f : 0.0f;
  if (!inr) loss = 0.0f;

  // In-wave reductions on the matrix pipe (EXEC is all-ones here: no early
  // returns above, only converged predication).
  const float wsum = wave_sum_f32(loss);
  const float wcnt = wave_count_wmma(vld);
  if ((threadIdx.x & 31) == 0) {
    atomicAdd(&ws[0], wsum);
    atomicAdd(&ws[1], wcnt);
  }
}

__global__ void finalize_kernel(const float* __restrict__ ws,
                                float* __restrict__ out) {
  out[0] = ws[0] / fmaxf(ws[1], 1.0f);
}

extern "C" void kernel_launch(void* const* d_in, const int* in_sizes, int n_in,
                              void* d_out, int out_size, void* d_ws,
                              size_t ws_size, hipStream_t stream) {
  const float*         scores   = (const float*)d_in[0];
  const int*           rankings = (const int*)d_in[1];
  const unsigned char* mask     = (const unsigned char*)d_in[2];
  float* out = (float*)d_out;
  float* ws  = (float*)d_ws;

  const int B = in_sizes[1] / H_CONST;  // rankings is (B, H)

  ws_zero_kernel<<<1, 1, 0, stream>>>(ws);
  const int threads = 256;
  const int blocks  = (B + threads - 1) / threads;
  pl_loss_kernel<<<blocks, threads, 0, stream>>>(scores, rankings, mask, ws, B);
  finalize_kernel<<<1, 1, 0, stream>>>(ws, out);
}
